// DifferentialAttention_13864154432312
// MI455X (gfx1250) — matrix operand order repaired and checked by reading, not compiled, not run on hardware
//
#include <hip/hip_runtime.h>
#include <hip/hip_bf16.h>

typedef __bf16 bf16;
typedef __attribute__((ext_vector_type(16))) __bf16 v16bf;
typedef __attribute__((ext_vector_type(8)))  float  v8f;

#define DIM   1024
#define NSEQ  2048
#define NHEAD 16
#define HD    64
#define NROWS 4096           /* B*N */
#define FDIM  5120           /* 5*DIM */
#define SCALE 0.125f         /* 64^-0.5 */

// ---------- helpers -------------------------------------------------------

__device__ __forceinline__ bf16 f2bf(float f) {
    unsigned u = __float_as_uint(f);
    unsigned r = u + 0x7FFFu + ((u >> 16) & 1u);   // round-to-nearest-even
    unsigned short h = (unsigned short)(r >> 16);
    return __builtin_bit_cast(bf16, h);
}

__device__ __forceinline__ v8f wmma_bf16(v16bf a, v16bf b, v8f c) {
    return __builtin_amdgcn_wmma_f32_16x16x32_bf16(
        false, a, false, b, (short)0, c, false, false);
}

// A fragment: 16x32 bf16, rows m0..m0+15, cols k0..k0+31 of row-major matrix.
// ISA layout: lane<16 -> row=lane, K 0..7 (V0-3) + K 16..23 (V4-7);
//             lane>=16 -> row=lane-16, K 8..15 + 24..31.
__device__ __forceinline__ v16bf ld_fragA_g(const bf16* __restrict__ base,
                                            int stride, int m0, int k0) {
    int lane = threadIdx.x & 31;
    int hf = lane >> 4, r = lane & 15;
    const bf16* p = base + (size_t)(m0 + r) * stride + (k0 + hf * 8);
    union { v16bf v; float4 q[2]; } u;
    u.q[0] = *(const float4*)(p);
    u.q[1] = *(const float4*)(p + 16);
    return u.v;
}

// Same A layout but from LDS with runtime stride (P tile row-major).
__device__ __forceinline__ v16bf ld_fragA_lds(const bf16* p0, int stride, int k0) {
    int lane = threadIdx.x & 31;
    int hf = lane >> 4, r = lane & 15;
    const bf16* p = p0 + r * stride + k0 + hf * 8;
    union { v16bf v; float4 q[2]; } u;
    u.q[0] = *(const float4*)(p);
    u.q[1] = *(const float4*)(p + 16);
    return u.v;
}

// B fragment: 32x16 bf16 taken as rows n0..n0+15 of an N-major matrix
// (i.e. B^T row-major, contiguous along K).
// ISA layout: lane<16 -> N=lane, K 0..7 + 16..23; lane>=16 -> K 8..15 + 24..31.
__device__ __forceinline__ v16bf ld_fragB_g(const bf16* __restrict__ base,
                                            int stride, int n0, int k0) {
    int lane = threadIdx.x & 31;
    int hf = lane >> 4, c = lane & 15;
    const bf16* p = base + (size_t)(n0 + c) * stride + (k0 + hf * 8);
    union { v16bf v; float4 q[2]; } u;
    u.q[0] = *(const float4*)(p);
    u.q[1] = *(const float4*)(p + 16);
    return u.v;
}

// ---------- small kernels -------------------------------------------------

__global__ void cvt_f32_bf16(const float* __restrict__ in,
                             bf16* __restrict__ out, int n) {
    int i = blockIdx.x * blockDim.x + threadIdx.x;
    if (i < n) out[i] = f2bf(in[i]);
}

__global__ void lambda_kernel(const float* lq1, const float* lk1,
                              const float* lq2, const float* lk2,
                              float* lam) {
    int t = threadIdx.x;                        // 32 threads, 64 elements
    float s1 = lq1[t] * lk1[t] + lq1[t + 32] * lk1[t + 32];
    float s2 = lq2[t] * lk2[t] + lq2[t + 32] * lk2[t + 32];
    #pragma unroll
    for (int off = 16; off > 0; off >>= 1) {
        s1 += __shfl_xor(s1, off, 32);
        s2 += __shfl_xor(s2, off, 32);
    }
    if (t == 0) *lam = __expf(s1) - __expf(s2) + 0.8f;
}

// RMSNorm over C=1024, scale by (1-0.8), emit bf16 for the proj GEMM.
__global__ __launch_bounds__(256) void rmsnorm_kernel(
        const float* __restrict__ in, bf16* __restrict__ out) {
    __shared__ float red[256];
    int row = blockIdx.x;
    const float* p = in + (size_t)row * DIM;
    float v[4]; float s = 0.f;
    #pragma unroll
    for (int i = 0; i < 4; i++) {
        v[i] = p[threadIdx.x + i * 256];
        s += v[i] * v[i];
    }
    red[threadIdx.x] = s;
    __syncthreads();
    for (int off = 128; off > 0; off >>= 1) {
        if (threadIdx.x < off) red[threadIdx.x] += red[threadIdx.x + off];
        __syncthreads();
    }
    float r = rsqrtf(red[0] * (1.0f / DIM) + 1e-5f) * 0.2f;
    bf16* q = out + (size_t)row * DIM;
    #pragma unroll
    for (int i = 0; i < 4; i++)
        q[threadIdx.x + i * 256] = f2bf(v[i] * r);
}

// ---------- double-buffered 32x64 wave-tile WMMA core ---------------------
// acc[i][t] += A[m0+16i : +16, :] x B[n0+16t : +16, :]^T over K = DIM.
// Next k-step fragments are loaded into fresh registers before the current
// WMMAs consume theirs, so loads overlap matrix ops (partial loadcnt waits).

__device__ __forceinline__ void gemm_acc_32x64(
        const bf16* __restrict__ A, const bf16* __restrict__ B,
        int m0, int n0, v8f acc[2][4]) {
    v16bf a0 = ld_fragA_g(A, DIM, m0,      0);
    v16bf a1 = ld_fragA_g(A, DIM, m0 + 16, 0);
    v16bf b0 = ld_fragB_g(B, DIM, n0,      0);
    v16bf b1 = ld_fragB_g(B, DIM, n0 + 16, 0);
    v16bf b2 = ld_fragB_g(B, DIM, n0 + 32, 0);
    v16bf b3 = ld_fragB_g(B, DIM, n0 + 48, 0);
    for (int k0 = 0; k0 < DIM; k0 += 32) {
        v16bf na0 = a0, na1 = a1, nb0 = b0, nb1 = b1, nb2 = b2, nb3 = b3;
        int kn = k0 + 32;
        if (kn < DIM) {
            na0 = ld_fragA_g(A, DIM, m0,      kn);
            na1 = ld_fragA_g(A, DIM, m0 + 16, kn);
            nb0 = ld_fragB_g(B, DIM, n0,      kn);
            nb1 = ld_fragB_g(B, DIM, n0 + 16, kn);
            nb2 = ld_fragB_g(B, DIM, n0 + 32, kn);
            nb3 = ld_fragB_g(B, DIM, n0 + 48, kn);
        }
        acc[0][0] = wmma_bf16(a0, b0, acc[0][0]);
        acc[0][1] = wmma_bf16(a0, b1, acc[0][1]);
        acc[0][2] = wmma_bf16(a0, b2, acc[0][2]);
        acc[0][3] = wmma_bf16(a0, b3, acc[0][3]);
        acc[1][0] = wmma_bf16(a1, b0, acc[1][0]);
        acc[1][1] = wmma_bf16(a1, b1, acc[1][1]);
        acc[1][2] = wmma_bf16(a1, b2, acc[1][2]);
        acc[1][3] = wmma_bf16(a1, b3, acc[1][3]);
        a0 = na0; a1 = na1; b0 = nb0; b1 = nb1; b2 = nb2; b3 = nb3;
    }
}

// ---------- QKV GEMM: (4096x1024)x(1024x5120) with scatter epilogue -------

__global__ __launch_bounds__(256) void gemm_qkv_kernel(
        const bf16* __restrict__ xb, const bf16* __restrict__ wb,
        bf16* __restrict__ q1, bf16* __restrict__ q2,
        bf16* __restrict__ k1, bf16* __restrict__ k2,
        bf16* __restrict__ vT) {
    int wave = blockIdx.x * 8 + (threadIdx.x >> 5);
    const int NT = FDIM / 64;                  // 80
    int m0 = (wave / NT) * 32;
    int n0 = (wave % NT) * 64;
    v8f z = {};
    v8f acc[2][4] = {{z, z, z, z}, {z, z, z, z}};
    gemm_acc_32x64(xb, wb, m0, n0, acc);

    int lane = threadIdx.x & 31, hf = lane >> 4, cl = lane & 15;
    #pragma unroll
    for (int i = 0; i < 2; i++) {
        #pragma unroll
        for (int t = 0; t < 4; t++) {
            #pragma unroll
            for (int j = 0; j < 8; j++) {
                int row = m0 + i * 16 + j + hf * 8;    // [0,4096)
                int f   = n0 + t * 16 + cl;            // [0,5120)
                int bb = row >> 11, n = row & (NSEQ - 1);
                int five = f >> 10, h = (f >> 6) & 15, d = f & 63;
                int bh = bb * NHEAD + h;
                bf16 val = f2bf(acc[i][t][j]);
                if (five == 4) {               // V stored transposed (bh,hd,N)
                    vT[(size_t)bh * (HD * NSEQ) + (size_t)d * NSEQ + n] = val;
                } else {
                    bf16* dst = (five == 0) ? q1 : (five == 1) ? q2
                              : (five == 2) ? k1 : k2;
                    dst[(size_t)bh * (NSEQ * HD) + (size_t)n * HD + d] = val;
                }
            }
        }
    }
}

// ---------- dual flash attention (64-key blocks) --------------------------
// Wave = one 16-row Q tile of one (b,h). Per 64-key block:
//   16 WMMA scores (4 key tiles x 2 hd halves x 2 attns),
//   online softmax over 4 local values per row (halved shuffle cost/key),
//   P -> LDS (bf16, 16x64) -> two A fragments, 16 WMMA for PV.
// Next block's K/V rows prefetched (global_prefetch_b8) during softmax.

__device__ __forceinline__ void softmax_update(
        v8f s[4], float* m, float* l, v8f o[4],
        bf16* pbuf, int hf, int cl) {
    #pragma unroll
    for (int j = 0; j < 8; j++) {
        float v0 = s[0][j] * SCALE, v1 = s[1][j] * SCALE;
        float v2 = s[2][j] * SCALE, v3 = s[3][j] * SCALE;
        float mx = fmaxf(fmaxf(v0, v1), fmaxf(v2, v3));
        #pragma unroll
        for (int off = 1; off < 16; off <<= 1)
            mx = fmaxf(mx, __shfl_xor(mx, off, 32));
        float mn = fmaxf(m[j], mx);
        float alpha = __expf(m[j] - mn);
        float p0 = __expf(v0 - mn), p1 = __expf(v1 - mn);
        float p2 = __expf(v2 - mn), p3 = __expf(v3 - mn);
        float rs = (p0 + p1) + (p2 + p3);
        #pragma unroll
        for (int off = 1; off < 16; off <<= 1)
            rs += __shfl_xor(rs, off, 32);
        l[j] = l[j] * alpha + rs;
        m[j] = mn;
        o[0][j] *= alpha; o[1][j] *= alpha;
        o[2][j] *= alpha; o[3][j] *= alpha;
        int r = (j + hf * 8) * 64 + cl;
        pbuf[r]      = f2bf(p0);
        pbuf[r + 16] = f2bf(p1);
        pbuf[r + 32] = f2bf(p2);
        pbuf[r + 48] = f2bf(p3);
    }
}

__global__ __launch_bounds__(256) void attn_kernel(
        const bf16* __restrict__ q1, const bf16* __restrict__ q2,
        const bf16* __restrict__ k1, const bf16* __restrict__ k2,
        const bf16* __restrict__ vT, const float* __restrict__ lambda_p,
        float* __restrict__ out) {
    __shared__ bf16 lds[8][2][16 * 64];        // per-wave P tiles, 32 KB
    int wid  = threadIdx.x >> 5;
    int lane = threadIdx.x & 31;
    int hf = lane >> 4, cl = lane & 15;
    int gw = blockIdx.x * 8 + wid;             // [0, 32*128)
    int bh = gw >> 7;                          // (b,h)
    int qr = (gw & 127) * 16;                  // q-row base

    const bf16* q1h = q1 + (size_t)bh * NSEQ * HD;
    const bf16* q2h = q2 + (size_t)bh * NSEQ * HD;
    const bf16* k1h = k1 + (size_t)bh * NSEQ * HD;
    const bf16* k2h = k2 + (size_t)bh * NSEQ * HD;
    const bf16* vTh = vT + (size_t)bh * HD * NSEQ;

    v16bf aq1_0 = ld_fragA_g(q1h, HD, qr, 0);
    v16bf aq1_1 = ld_fragA_g(q1h, HD, qr, 32);
    v16bf aq2_0 = ld_fragA_g(q2h, HD, qr, 0);
    v16bf aq2_1 = ld_fragA_g(q2h, HD, qr, 32);

    v8f z = {};
    v8f o1[4] = {z, z, z, z};
    v8f o2[4] = {z, z, z, z};
    float m1[8], l1[8], m2[8], l2[8];
    #pragma unroll
    for (int j = 0; j < 8; j++) {
        m1[j] = -3.0e38f; m2[j] = -3.0e38f; l1[j] = 0.f; l2[j] = 0.f;
    }

    for (int kb = 0; kb < NSEQ; kb += 64) {
        // ---- scores: S = Q K^T, 16x64 per attention ----
        v8f s1[4], s2[4];
        #pragma unroll
        for (int u = 0; u < 4; u++) {
            int kt = kb + u * 16;
            v8f t1 = z, t2 = z;
            t1 = wmma_bf16(aq1_0, ld_fragB_g(k1h, HD, kt, 0),  t1);
            t1 = wmma_bf16(aq1_1, ld_fragB_g(k1h, HD, kt, 32), t1);
            t2 = wmma_bf16(aq2_0, ld_fragB_g(k2h, HD, kt, 0),  t2);
            t2 = wmma_bf16(aq2_1, ld_fragB_g(k2h, HD, kt, 32), t2);
            s1[u] = t1; s2[u] = t2;
        }

        // prefetch next key block while softmax runs (global_prefetch_b8)
        if (kb + 64 < NSEQ) {
            __builtin_prefetch(k1h + (size_t)(kb + 64 + lane) * HD, 0, 0);
            __builtin_prefetch(k2h + (size_t)(kb + 64 + lane) * HD, 0, 0);
            __builtin_prefetch(vTh + (size_t)lane * NSEQ + kb + 64, 0, 0);
            __builtin_prefetch(vTh + (size_t)(lane + 32) * NSEQ + kb + 64, 0, 0);
        }

        softmax_update(s1, m1, l1, o1, &lds[wid][0][0], hf, cl);
        softmax_update(s2, m2, l2, o2, &lds[wid][1][0], hf, cl);

        // wave-private LDS bounce: wait for our ds stores before reloading
        asm volatile("s_wait_dscnt 0" ::: "memory");

        v16bf ap1_0 = ld_fragA_lds(&lds[wid][0][0], 64, 0);
        v16bf ap1_1 = ld_fragA_lds(&lds[wid][0][0], 64, 32);
        v16bf ap2_0 = ld_fragA_lds(&lds[wid][1][0], 64, 0);
        v16bf ap2_1 = ld_fragA_lds(&lds[wid][1][0], 64, 32);

        // ---- PV: o += P (16x64) x V (64 x 64) ----
        #pragma unroll
        for (int t = 0; t < 4; t++) {
            v16bf bv0 = ld_fragB_g(vTh, NSEQ, t * 16, kb);
            v16bf bv1 = ld_fragB_g(vTh, NSEQ, t * 16, kb + 32);
            o1[t] = wmma_bf16(ap1_0, bv0, o1[t]);
            o1[t] = wmma_bf16(ap1_1, bv1, o1[t]);
            o2[t] = wmma_bf16(ap2_0, bv0, o2[t]);
            o2[t] = wmma_bf16(ap2_1, bv1, o2[t]);
        }
    }

    float lam = *lambda_p;
    int b = bh >> 4, h = bh & 15;
    #pragma unroll
    for (int j = 0; j < 8; j++) {
        float r1 = 1.0f / l1[j];
        float r2 = lam / l2[j];
        int row = qr + j + hf * 8;
        float* dst = out + ((size_t)b * NSEQ + row) * DIM + h * HD + cl;
        #pragma unroll
        for (int t = 0; t < 4; t++)
            dst[t * 16] = o1[t][j] * r1 - o2[t][j] * r2;
    }
}

// ---------- projection GEMM: (4096x1024)x(1024x1024) + bias -> d_out ------

__global__ __launch_bounds__(256) void gemm_proj_kernel(
        const bf16* __restrict__ yb, const bf16* __restrict__ pw,
        const float* __restrict__ bias, float* __restrict__ out) {
    int wave = blockIdx.x * 8 + (threadIdx.x >> 5);
    const int NT = DIM / 64;                   // 16
    int m0 = (wave / NT) * 32;
    int n0 = (wave % NT) * 64;
    v8f z = {};
    v8f acc[2][4] = {{z, z, z, z}, {z, z, z, z}};
    gemm_acc_32x64(yb, pw, m0, n0, acc);

    int lane = threadIdx.x & 31, hf = lane >> 4, cl = lane & 15;
    #pragma unroll
    for (int i = 0; i < 2; i++) {
        #pragma unroll
        for (int t = 0; t < 4; t++) {
            int f = n0 + t * 16 + cl;
            float bv = bias[f];
            #pragma unroll
            for (int j = 0; j < 8; j++) {
                int row = m0 + i * 16 + j + hf * 8;
                out[(size_t)row * DIM + f] = acc[i][t][j] + bv;
            }
        }
    }
}

// ---------- launch --------------------------------------------------------

extern "C" void kernel_launch(void* const* d_in, const int* in_sizes, int n_in,
                              void* d_out, int out_size, void* d_ws, size_t ws_size,
                              hipStream_t stream) {
    const float* x      = (const float*)d_in[0];
    const float* qkv_w  = (const float*)d_in[1];
    const float* proj_w = (const float*)d_in[2];
    const float* proj_b = (const float*)d_in[3];
    const float* lq1    = (const float*)d_in[4];
    const float* lk1    = (const float*)d_in[5];
    const float* lq2    = (const float*)d_in[6];
    const float* lk2    = (const float*)d_in[7];
    float* out = (float*)d_out;

    char* ws = (char*)d_ws;
    const size_t SZ_X   = (size_t)NROWS * DIM * 2;       // 8 MB  bf16
    const size_t SZ_QW  = (size_t)FDIM  * DIM * 2;       // 10 MB bf16
    const size_t SZ_PW  = (size_t)DIM   * DIM * 2;       // 2 MB  bf16
    const size_t SZ_HD  = (size_t)32 * NSEQ * HD * 2;    // 8 MB  bf16 (per q/k/v)
    const size_t SZ_AO  = (size_t)NROWS * DIM * 4;       // 16 MB f32

    size_t off = 0;
    bf16*  xb  = (bf16*)(ws + off); off += SZ_X;
    bf16*  qwb = (bf16*)(ws + off); off += SZ_QW;
    bf16*  pwb = (bf16*)(ws + off); off += SZ_PW;
    bf16*  q1  = (bf16*)(ws + off); off += SZ_HD;
    bf16*  q2  = (bf16*)(ws + off); off += SZ_HD;
    bf16*  k1  = (bf16*)(ws + off); off += SZ_HD;
    bf16*  k2  = (bf16*)(ws + off); off += SZ_HD;
    bf16*  vT  = (bf16*)(ws + off); off += SZ_HD;
    float* ao  = (float*)(ws + off); off += SZ_AO;
    bf16*  yb  = (bf16*)(ws + off); off += SZ_X;
    float* lam = (float*)(ws + off); off += 256;
    (void)ws_size; (void)in_sizes; (void)n_in; (void)out_size;

    // 1. bf16 conversions
    cvt_f32_bf16<<<(NROWS * DIM) / 256, 256, 0, stream>>>(x, xb, NROWS * DIM);
    cvt_f32_bf16<<<(FDIM * DIM) / 256, 256, 0, stream>>>(qkv_w, qwb, FDIM * DIM);
    cvt_f32_bf16<<<(DIM * DIM) / 256, 256, 0, stream>>>(proj_w, pwb, DIM * DIM);

    // 2. lambda_full scalar
    lambda_kernel<<<1, 32, 0, stream>>>(lq1, lk1, lq2, lk2, lam);

    // 3. QKV projection (WMMA) with q/k/v scatter (V transposed)
    //    waves = (4096/32)*(5120/64) = 10240 -> 1280 blocks of 8 waves
    gemm_qkv_kernel<<<1280, 256, 0, stream>>>(xb, qwb, q1, q2, k1, k2, vT);

    // 4. dual flash attention: waves = 32*(2048/16) = 4096 -> 512 blocks
    attn_kernel<<<512, 256, 0, stream>>>(q1, q2, k1, k2, vT, lam, ao);

    // 5. RMSNorm + 0.2 scale -> bf16
    rmsnorm_kernel<<<NROWS, 256, 0, stream>>>(ao, yb);

    // 6. output projection (WMMA) + bias
    //    waves = (4096/32)*(1024/64) = 2048 -> 256 blocks
    gemm_proj_kernel<<<256, 256, 0, stream>>>(yb, pwb, proj_b, out);
}
